// Pointnet_fp_module_33071248179394
// MI455X (gfx1250) — compile-verified, matrix operand order checked by reading, not run on hardware
//
#include <hip/hip_runtime.h>
#include <hip/hip_bf16.h>

typedef __bf16 bf16_t;
typedef __attribute__((ext_vector_type(16))) bf16_t v16bf;
typedef __attribute__((ext_vector_type(8)))  float  v8f;

// ---------------------------------------------------------------------------
// Fragment helpers.
// CDNA5 16-bit A/B WMMA layout (ISA 7.12.2): lanes 0-15 hold K 0..7 and 16..23
// (halves 0..7 / 8..15), lanes 16-31 hold K 8..15 and 24..31.  Both chunks are
// 16 contiguous bytes in a row-major (row, K) matrix -> two b128 loads.
// ---------------------------------------------------------------------------
union Frag32 { uint4 u[2]; v16bf v; };

__device__ inline v16bf load_frag_bf16(const bf16_t* __restrict__ rowp, int k0) {
  Frag32 f;
  f.u[0] = *reinterpret_cast<const uint4*>(rowp + k0);
  f.u[1] = *reinterpret_cast<const uint4*>(rowp + k0 + 16);
  return f.v;
}

// Build an A fragment from fp32 y1 with BN1 affine + ReLU fused, cast to bf16.
__device__ inline v16bf make_bn_frag(const float* __restrict__ rowp, int k0,
                                     const float* __restrict__ sa,
                                     const float* __restrict__ sc) {
  v16bf a;
#pragma unroll
  for (int h = 0; h < 16; ++h) {
    int k = k0 + (h < 8 ? h : h + 8);   // halves 8..15 -> K + 16
    float v = fmaf(rowp[k], sa[k], sc[k]);
    v = fmaxf(v, 0.0f);
    a[h] = (bf16_t)v;
  }
  return a;
}

__device__ inline void top3_insert(float d, int idx,
                                   float& d0, int& j0,
                                   float& d1, int& j1,
                                   float& d2, int& j2) {
  if (d < d2) {
    if (d < d1) {
      d2 = d1; j2 = j1;
      if (d < d0) { d1 = d0; j1 = j0; d0 = d; j0 = idx; }
      else        { d1 = d;  j1 = idx; }
    } else { d2 = d; j2 = idx; }
  }
}

// ---------------------------------------------------------------------------
// Kernel 0: cast weights to bf16, zero BN stat accumulators.
// ---------------------------------------------------------------------------
__global__ __launch_bounds__(256) void prep_kernel(
    const float* __restrict__ W1, const float* __restrict__ W2,
    bf16_t* __restrict__ w1b, bf16_t* __restrict__ w2b,
    float* __restrict__ stats) {
  int t = blockIdx.x * 256 + threadIdx.x;
  if (t < 196608) w1b[t] = (bf16_t)W1[t];          // 256*768
  int t2 = t - 196608;
  if (t2 >= 0 && t2 < 65536) w2b[t2] = (bf16_t)W2[t2];  // 256*256
  if (t < 2048) stats[t] = 0.0f;
}

// ---------------------------------------------------------------------------
// Kernel 1: three_nn + inverse-distance interpolation + concat -> bf16 feat0.
// One wave per query point (8 points / 256-thread block).
// ---------------------------------------------------------------------------
__global__ __launch_bounds__(256) void three_nn_interp_kernel(
    const float* __restrict__ xyz1, const float* __restrict__ xyz2,
    const float* __restrict__ points1, const float* __restrict__ points2,
    bf16_t* __restrict__ feat0) {
  const int tid = threadIdx.x, lane = tid & 31, w = tid >> 5;
  const int pbase = blockIdx.x * 8;     // 4096 % 8 == 0 -> one batch per block
  const int p = pbase + w;
  const int bb = p >> 12;               // p / 4096

  __shared__ float s_w[8][3];
  __shared__ int   s_idx[8][3];

  const float px = xyz1[p * 3 + 0];
  const float py = xyz1[p * 3 + 1];
  const float pz = xyz1[p * 3 + 2];

  float d0 = 3e38f, d1 = 3e38f, d2 = 3e38f;
  int   j0 = 0, j1 = 0, j2 = 0;
  const float* x2 = xyz2 + (size_t)bb * 1024 * 3;
#pragma unroll 4
  for (int m = lane; m < 1024; m += 32) {
    float dx = x2[m * 3 + 0] - px;
    float dy = x2[m * 3 + 1] - py;
    float dz = x2[m * 3 + 2] - pz;
    float d = dx * dx + dy * dy + dz * dz;
    top3_insert(d, m, d0, j0, d1, j1, d2, j2);
  }
  // butterfly merge of per-lane top3 across the wave32
  for (int off = 16; off >= 1; off >>= 1) {
    float e0 = __shfl_xor(d0, off), e1 = __shfl_xor(d1, off), e2 = __shfl_xor(d2, off);
    int   k0 = __shfl_xor(j0, off), k1 = __shfl_xor(j1, off), k2 = __shfl_xor(j2, off);
    top3_insert(e0, k0, d0, j0, d1, j1, d2, j2);
    top3_insert(e1, k1, d0, j0, d1, j1, d2, j2);
    top3_insert(e2, k2, d0, j0, d1, j1, d2, j2);
  }
  if (lane == 0) {
    // Hardtanh(0, 1e-10) on squared dists, then inverse-distance weights
    float c0 = fminf(d0, 1e-10f), c1 = fminf(d1, 1e-10f), c2 = fminf(d2, 1e-10f);
    float i0 = 1.0f / c0, i1 = 1.0f / c1, i2 = 1.0f / c2;
    float s = i0 + i1 + i2;
    s_w[w][0] = i0 / s; s_w[w][1] = i1 / s; s_w[w][2] = i2 / s;
    s_idx[w][0] = j0; s_idx[w][1] = j1; s_idx[w][2] = j2;
  }
  __syncthreads();

  const float* p2 = points2 + (size_t)bb * 1024 * 512;
#pragma unroll
  for (int pp = 0; pp < 8; ++pp) {
    int q = pbase + pp;
    float w0 = s_w[pp][0], w1v = s_w[pp][1], w2v = s_w[pp][2];
    const float* r0 = p2 + (size_t)s_idx[pp][0] * 512;
    const float* r1 = p2 + (size_t)s_idx[pp][1] * 512;
    const float* r2 = p2 + (size_t)s_idx[pp][2] * 512;
    bf16_t* fo = feat0 + (size_t)q * 768;
    for (int c = tid; c < 512; c += 256)
      fo[c] = (bf16_t)(w0 * r0[c] + w1v * r1[c] + w2v * r2[c]);
    fo[512 + tid] = (bf16_t)points1[(size_t)q * 256 + tid];  // concat points1
  }
}

// ---------------------------------------------------------------------------
// Kernel 2: GEMM1 (65536x768 @ 768x256) bf16 WMMA + bias + BN1 stats.
// Block tile 128(M) x 64(N); 8 waves of 32x32 (2x2 v_wmma_f32_16x16x32_bf16).
// ---------------------------------------------------------------------------
__global__ __launch_bounds__(256) void gemm1_kernel(
    const bf16_t* __restrict__ A,   // (65536,768) bf16
    const bf16_t* __restrict__ W,   // (256,768) bf16 row-major (out,in)
    const float* __restrict__ bias, // b1
    float* __restrict__ Y,          // (65536,256) fp32 pre-BN
    float* __restrict__ gsum, float* __restrict__ gsq) {
  const int K = 768, N = 256;
  const int tid = threadIdx.x, lane = tid & 31, w = tid >> 5;
  const int wm = w & 3, wn = w >> 2;
  const int blockM = blockIdx.x * 128, blockN = blockIdx.y * 64;
  const int mt = blockM + wm * 32, nt = blockN + wn * 32;
  const int lrow = lane & 15;
  const int klane = (lane & 16) ? 8 : 0;   // K half-select per ISA layout
  const int hi = (lane & 16) ? 8 : 0;      // C/D row offset for upper lanes

  __shared__ float s_sum[64], s_sq[64];
  if (tid < 64) { s_sum[tid] = 0.0f; s_sq[tid] = 0.0f; }
  __syncthreads();

  v8f acc[2][2] = {};
  const bf16_t* ar0 = A + (size_t)(mt + lrow) * K;
  const bf16_t* ar1 = A + (size_t)(mt + 16 + lrow) * K;
  const bf16_t* br0 = W + (size_t)(nt + lrow) * K;
  const bf16_t* br1 = W + (size_t)(nt + 16 + lrow) * K;

  for (int kk = 0; kk < K; kk += 32) {
    if (kk + 32 < K) {                       // global_prefetch_b8 next A chunk
      __builtin_prefetch(ar0 + kk + 32 + klane, 0, 1);
      __builtin_prefetch(ar1 + kk + 32 + klane, 0, 1);
    }
    v16bf af0 = load_frag_bf16(ar0, kk + klane);
    v16bf af1 = load_frag_bf16(ar1, kk + klane);
    v16bf bf0 = load_frag_bf16(br0, kk + klane);
    v16bf bf1 = load_frag_bf16(br1, kk + klane);
    acc[0][0] = __builtin_amdgcn_wmma_f32_16x16x32_bf16(false, af0, false, bf0, (short)0, acc[0][0], false, false);
    acc[0][1] = __builtin_amdgcn_wmma_f32_16x16x32_bf16(false, af0, false, bf1, (short)0, acc[0][1], false, false);
    acc[1][0] = __builtin_amdgcn_wmma_f32_16x16x32_bf16(false, af1, false, bf0, (short)0, acc[1][0], false, false);
    acc[1][1] = __builtin_amdgcn_wmma_f32_16x16x32_bf16(false, af1, false, bf1, (short)0, acc[1][1], false, false);
  }

#pragma unroll
  for (int mi = 0; mi < 2; ++mi)
#pragma unroll
    for (int nj = 0; nj < 2; ++nj) {
      int n = nt + nj * 16 + lrow;
      float bv = bias[n];
      float ps = 0.0f, pq = 0.0f;
#pragma unroll
      for (int r = 0; r < 8; ++r) {
        float v = acc[mi][nj][r] + bv;
        ps += v; pq += v * v;
        int m = mt + mi * 16 + hi + r;
        Y[(size_t)m * N + n] = v;
      }
      ps += __shfl_xor(ps, 16);              // lanes l / l+16 share channel n
      pq += __shfl_xor(pq, 16);
      if (lane < 16) {
        atomicAdd(&s_sum[wn * 32 + nj * 16 + lane], ps);
        atomicAdd(&s_sq [wn * 32 + nj * 16 + lane], pq);
      }
    }
  __syncthreads();
  if (tid < 64) {
    atomicAdd(&gsum[blockN + tid], s_sum[tid]);
    atomicAdd(&gsq [blockN + tid], s_sq[tid]);
  }
}

// ---------------------------------------------------------------------------
// Kernel 3/5: finalize BN statistics -> per-channel scale a, shift c.
// ---------------------------------------------------------------------------
__global__ __launch_bounds__(256) void finalize_stats_kernel(
    const float* __restrict__ sum, const float* __restrict__ sq,
    const float* __restrict__ g, const float* __restrict__ beta,
    float* __restrict__ a, float* __restrict__ c, float invM) {
  int n = threadIdx.x;
  float mean = sum[n] * invM;
  float var = sq[n] * invM - mean * mean;
  float s = g[n] * rsqrtf(var + 1e-5f);
  a[n] = s;
  c[n] = beta[n] - mean * s;
}

// ---------------------------------------------------------------------------
// Kernel 4: fused BN1+ReLU -> GEMM2 (65536x256 @ 256x256) + bias + BN2 stats,
// LDS tile transpose for coalesced (b, c, n) output store.
// ---------------------------------------------------------------------------
__global__ __launch_bounds__(256) void gemm2_kernel(
    const float* __restrict__ Yin,   // (65536,256) pre-BN1
    const bf16_t* __restrict__ W,    // (256,256) bf16
    const float* __restrict__ bias,  // b2
    const float* __restrict__ a1, const float* __restrict__ c1,
    float* __restrict__ out,         // (16,256,4096) raw pre-BN2
    float* __restrict__ gsum, float* __restrict__ gsq) {
  const int K = 256;
  __shared__ float s_a[256], s_c[256];
  __shared__ float s_sum[64], s_sq[64];
  __shared__ float s_tile[128][65];   // +1 pad: bank-conflict-free transpose

  const int tid = threadIdx.x, lane = tid & 31, w = tid >> 5;
  const int wm = w & 3, wn = w >> 2;
  const int blockM = blockIdx.x * 128, blockN = blockIdx.y * 64;
  const int mt = blockM + wm * 32, nt = blockN + wn * 32;
  const int lrow = lane & 15;
  const int klane = (lane & 16) ? 8 : 0;
  const int hi = (lane & 16) ? 8 : 0;

  s_a[tid] = a1[tid];
  s_c[tid] = c1[tid];
  if (tid < 64) { s_sum[tid] = 0.0f; s_sq[tid] = 0.0f; }
  __syncthreads();

  v8f acc[2][2] = {};
  const float*  ar0 = Yin + (size_t)(mt + lrow) * K;
  const float*  ar1 = Yin + (size_t)(mt + 16 + lrow) * K;
  const bf16_t* br0 = W + (size_t)(nt + lrow) * K;
  const bf16_t* br1 = W + (size_t)(nt + 16 + lrow) * K;

  for (int kk = 0; kk < K; kk += 32) {
    v16bf af0 = make_bn_frag(ar0, kk + klane, s_a, s_c);
    v16bf af1 = make_bn_frag(ar1, kk + klane, s_a, s_c);
    v16bf bf0 = load_frag_bf16(br0, kk + klane);
    v16bf bf1 = load_frag_bf16(br1, kk + klane);
    acc[0][0] = __builtin_amdgcn_wmma_f32_16x16x32_bf16(false, af0, false, bf0, (short)0, acc[0][0], false, false);
    acc[0][1] = __builtin_amdgcn_wmma_f32_16x16x32_bf16(false, af0, false, bf1, (short)0, acc[0][1], false, false);
    acc[1][0] = __builtin_amdgcn_wmma_f32_16x16x32_bf16(false, af1, false, bf0, (short)0, acc[1][0], false, false);
    acc[1][1] = __builtin_amdgcn_wmma_f32_16x16x32_bf16(false, af1, false, bf1, (short)0, acc[1][1], false, false);
  }

#pragma unroll
  for (int mi = 0; mi < 2; ++mi)
#pragma unroll
    for (int nj = 0; nj < 2; ++nj) {
      int nloc = wn * 32 + nj * 16 + lrow;
      float bv = bias[blockN + nloc];
      float ps = 0.0f, pq = 0.0f;
#pragma unroll
      for (int r = 0; r < 8; ++r) {
        float v = acc[mi][nj][r] + bv;
        ps += v; pq += v * v;
        s_tile[wm * 32 + mi * 16 + hi + r][nloc] = v;
      }
      ps += __shfl_xor(ps, 16);
      pq += __shfl_xor(pq, 16);
      if (lane < 16) {
        atomicAdd(&s_sum[wn * 32 + nj * 16 + lane], ps);
        atomicAdd(&s_sq [wn * 32 + nj * 16 + lane], pq);
      }
    }
  __syncthreads();
  if (tid < 64) {
    atomicAdd(&gsum[blockN + tid], s_sum[tid]);
    atomicAdd(&gsq [blockN + tid], s_sq[tid]);
  }

  // coalesced transposed store: out[bb][blockN+c][ibase+ml]
  const int bb = blockM >> 12;          // 4096 % 128 == 0
  const int ibase = blockM & 4095;
#pragma unroll
  for (int e = 0; e < 32; ++e) {
    int flat = tid + e * 256;
    int c = flat >> 7, ml = flat & 127;
    out[(size_t)bb * 256 * 4096 + (size_t)(blockN + c) * 4096 + ibase + ml] =
        s_tile[ml][c];
  }
}

// ---------------------------------------------------------------------------
// Kernel 6: in-place BN2 affine + ReLU over the (b, c, n) output, float4.
// ---------------------------------------------------------------------------
__global__ __launch_bounds__(256) void bn2_relu_kernel(
    float* __restrict__ out, const float* __restrict__ a2,
    const float* __restrict__ c2) {
  size_t f = ((size_t)blockIdx.x * 256 + threadIdx.x) * 4;
  int ch = (int)((f >> 12) & 255);
  float a = a2[ch], c = c2[ch];
  float4 v = *reinterpret_cast<float4*>(out + f);
  v.x = fmaxf(fmaf(v.x, a, c), 0.0f);
  v.y = fmaxf(fmaf(v.y, a, c), 0.0f);
  v.z = fmaxf(fmaf(v.z, a, c), 0.0f);
  v.w = fmaxf(fmaf(v.w, a, c), 0.0f);
  *reinterpret_cast<float4*>(out + f) = v;
}

// ---------------------------------------------------------------------------
extern "C" void kernel_launch(void* const* d_in, const int* in_sizes, int n_in,
                              void* d_out, int out_size, void* d_ws, size_t ws_size,
                              hipStream_t stream) {
  const float* xyz1    = (const float*)d_in[0];
  const float* xyz2    = (const float*)d_in[1];
  const float* points1 = (const float*)d_in[2];
  const float* points2 = (const float*)d_in[3];
  const float* W1      = (const float*)d_in[4];
  const float* b1      = (const float*)d_in[5];
  const float* g1      = (const float*)d_in[6];
  const float* beta1   = (const float*)d_in[7];
  const float* W2      = (const float*)d_in[8];
  const float* b2      = (const float*)d_in[9];
  const float* g2      = (const float*)d_in[10];
  const float* beta2   = (const float*)d_in[11];
  float* out = (float*)d_out;

  char* ws = (char*)d_ws;
  bf16_t* feat0 = (bf16_t*)ws;                    // 65536*768*2 = 100663296 B
  float*  y1    = (float*)(ws + 100663296);       // 65536*256*4 =  67108864 B
  bf16_t* w1b   = (bf16_t*)(ws + 167772160);      // 256*768*2   =    393216 B
  bf16_t* w2b   = (bf16_t*)(ws + 168165376);      // 256*256*2   =    131072 B
  float*  stats = (float*)(ws + 168296448);       // 2048 floats
  float* sum1 = stats,        *sq1 = stats + 256;
  float* aa1  = stats + 512,  *cc1 = stats + 768;
  float* sum2 = stats + 1024, *sq2 = stats + 1280;
  float* aa2  = stats + 1536, *cc2 = stats + 1792;

  const float invM = 1.0f / 65536.0f;

  prep_kernel<<<1024, 256, 0, stream>>>(W1, W2, w1b, w2b, stats);
  three_nn_interp_kernel<<<8192, 256, 0, stream>>>(xyz1, xyz2, points1, points2, feat0);
  gemm1_kernel<<<dim3(512, 4), 256, 0, stream>>>(feat0, w1b, b1, y1, sum1, sq1);
  finalize_stats_kernel<<<1, 256, 0, stream>>>(sum1, sq1, g1, beta1, aa1, cc1, invM);
  gemm2_kernel<<<dim3(512, 4), 256, 0, stream>>>(y1, w2b, b2, aa1, cc1, out, sum2, sq2);
  finalize_stats_kernel<<<1, 256, 0, stream>>>(sum2, sq2, g2, beta2, aa2, cc2, invM);
  bn2_relu_kernel<<<16384, 256, 0, stream>>>(out, aa2, cc2);
}
